// WeightedSAGEConv_1417339208119
// MI455X (gfx1250) — matrix operand order repaired and checked by reading, not compiled, not run on hardware
//
#include <hip/hip_runtime.h>
#include <hip/hip_bf16.h>
#include <cstdint>

#define N_SRC  100000
#define N_DST  25000
#define N_EDGE 300000
// D_IN = D_HID = D_OUT = 256 everywhere

typedef __attribute__((ext_vector_type(16))) __bf16        v16bf;
typedef __attribute__((ext_vector_type(4)))  __bf16        v4bf;
typedef __attribute__((ext_vector_type(8)))  float         v8f;
typedef __attribute__((ext_vector_type(4)))  unsigned int  v4u;

#if defined(__has_builtin)
#if __has_builtin(__builtin_amdgcn_global_load_async_to_lds_b128) && \
    __has_builtin(__builtin_amdgcn_s_wait_asynccnt)
#define HAVE_ASYNC_LDS 1
#endif
#endif
#ifndef HAVE_ASYNC_LDS
#define HAVE_ASYNC_LDS 0
#endif

#if HAVE_ASYNC_LDS
// builtin prototype (from hipcc diagnostic): (V4i AS1*, V4i AS3*, imm int, imm int)
typedef int v4i_vs __attribute__((vector_size(4 * sizeof(int))));
typedef __attribute__((address_space(1))) v4i_vs* as1_v4i;
typedef __attribute__((address_space(3))) v4i_vs* as3_v4i;

__device__ __forceinline__ void async_cp16(const void* g, void* l) {
  __builtin_amdgcn_global_load_async_to_lds_b128(
      (as1_v4i)(v4i_vs*)g, (as3_v4i)(v4i_vs*)l, 0, 0);
}
#endif

// ---------- helpers ----------
// native fptrunc f32 -> bf16 (RNE); backend picks hardware cvt when available
__device__ __forceinline__ unsigned short to_bf16_bits(float f) {
  __bf16 h = (__bf16)f;
  return __builtin_bit_cast(unsigned short, h);
}

struct Pair32 { v4u a, b; };

__device__ __forceinline__ v16bf ld_frag(const unsigned short* p0,
                                         const unsigned short* p1) {
  Pair32 t;
  t.a = *(const v4u*)p0;                     // 16B aligned LDS read
  t.b = *(const v4u*)p1;
  return __builtin_bit_cast(v16bf, t);
}

// ---------- zero init ----------
__global__ void zero_f32(float* __restrict__ p, int n) {
  int i = blockIdx.x * blockDim.x + threadIdx.x;
  if (i < n) p[i] = 0.f;
}

// ---------- pack weights: B f32 [K,256] -> Bt bf16 [256][K] (column-major) ----------
__global__ void pack_b_bf16(const float* __restrict__ B,
                            unsigned short* __restrict__ Bt, int K) {
  int k = blockIdx.x * blockDim.x + threadIdx.x;
  int n = blockIdx.y;
  if (k < K) Bt[(size_t)n * K + k] = to_bf16_bits(B[(size_t)k * 256 + n]);
}

// ---------- fused bf16-WMMA GEMM + bias + ReLU ----------
// C[M,256] = relu(A @ B + bias), B pre-packed bf16 col-major Bt[256][KDIM]
// MODE 0: A = A1 [M,KDIM]
// MODE 1: A = concat(A1[M,256]*rowscale[M], A2[M,256]), KDIM = 512
template<int KDIM, int MODE>
__global__ __launch_bounds__(128)
void gemm_bf16_relu(const float* __restrict__ A1, const float* __restrict__ A2,
                    const float* __restrict__ rowscale,
                    const unsigned short* __restrict__ Bt,
                    const float* __restrict__ bias,
                    float* __restrict__ C, int M)
{
  __shared__ __align__(16) unsigned short As[64][40]; // [m][k], +8 pad
  __shared__ __align__(16) unsigned short Bs[64][40]; // [n][k], +8 pad

  const int tid  = threadIdx.x;
  const int lane = tid & 31;
  const int wave = tid >> 5;
  const int kh   = lane >> 4;               // lane half (wave32)
  const int m0   = blockIdx.x * 64;
  const int n0   = blockIdx.y * 64;
  constexpr int NK = KDIM >> 5;

  v8f acc[4];
  const v8f vzero = {0.f,0.f,0.f,0.f,0.f,0.f,0.f,0.f};
  acc[0] = vzero; acc[1] = vzero; acc[2] = vzero; acc[3] = vzero;

  for (int kb = 0; kb < NK; ++kb) {
    // ---- B tile 64x32 bf16: async global->LDS copy (no cvt, no transpose) ----
#pragma unroll
    for (int i = 0; i < 2; ++i) {
      int c = tid * 2 + i;                  // 256 chunks of 16B
      int r = c >> 2;                       // 0..63  (n_local)
      int o = (c & 3) * 8;                  // bf16 element offset 0/8/16/24
      const unsigned short* g = Bt + (size_t)(n0 + r) * KDIM + (kb << 5) + o;
#if HAVE_ASYNC_LDS
      async_cp16(g, &Bs[r][o]);
#else
      *(v4u*)&Bs[r][o] = *(const v4u*)g;
#endif
    }

    // ---- A tile 64x32: f32 load + native cvt -> bf16 LDS ----
#pragma unroll
    for (int i = 0; i < 4; ++i) {
      int idx = tid + i * 128;              // 512 float4 chunks
      int row = idx >> 3;
      int c4  = (idx & 7) * 4;
      int g   = m0 + row;
      int kg  = (kb << 5) + c4;
      float4 v = make_float4(0.f, 0.f, 0.f, 0.f);
      if (g < M) {
        if constexpr (MODE == 0) {
          v = *(const float4*)(A1 + (size_t)g * KDIM + kg);
          if (kb + 1 < NK)                  // global_prefetch_b8 of next A chunk
            __builtin_prefetch(A1 + (size_t)g * KDIM + kg + 32, 0, 1);
        } else {
          if (kg < 256) {
            v = *(const float4*)(A1 + (size_t)g * 256 + kg);
            float s = rowscale[g];
            v.x *= s; v.y *= s; v.z *= s; v.w *= s;
          } else {
            v = *(const float4*)(A2 + (size_t)g * 256 + (kg - 256));
          }
        }
      }
      v4bf pk = { (__bf16)v.x, (__bf16)v.y, (__bf16)v.z, (__bf16)v.w };
      *(v4bf*)&As[row][c4] = pk;            // 8B LDS store (c4*2 bytes, 8B aligned)
    }

#if HAVE_ASYNC_LDS
    __builtin_amdgcn_s_wait_asynccnt(0);
#endif
    __syncthreads();

    // ---- WMMA: A frag per ISA 16-bit layout, B column-per-lane ----
    const int arow = wave * 16 + (lane & 15);
    v16bf afrag = ld_frag(&As[arow][kh * 8], &As[arow][kh * 8 + 16]);
#pragma unroll
    for (int nt = 0; nt < 4; ++nt) {
      const unsigned short* bp = &Bs[nt * 16 + (lane & 15)][kh * 8];
      v16bf bfrag = ld_frag(bp, bp + 16);
      acc[nt] = __builtin_amdgcn_wmma_f32_16x16x32_bf16(
          false, afrag, false, bfrag, (short)0, acc[nt], false, false);
    }
    __syncthreads();
  }

  // ---- epilogue: bias + ReLU, guard M ----
#pragma unroll
  for (int nt = 0; nt < 4; ++nt) {
    int col  = n0 + nt * 16 + (lane & 15);
    float bv = bias[col];
#pragma unroll
    for (int i = 0; i < 8; ++i) {
      int r = m0 + wave * 16 + kh * 8 + i;  // C/D layout: elem i -> M = i (+8 hi lanes)
      if (r < M) {
        float val = acc[nt][i] + bv;
        C[(size_t)r * 256 + col] = fmaxf(val, 0.f);
      }
    }
  }
}

// ---------- edge scatter: one wave per edge ----------
__global__ __launch_bounds__(256)
void edge_scatter(const float* __restrict__ nft, const float* __restrict__ wts,
                  const int* __restrict__ src_idx, const int* __restrict__ dst_idx,
                  float* __restrict__ nacc, float* __restrict__ wsum, int E)
{
  int wave = threadIdx.x >> 5;
  int lane = threadIdx.x & 31;
  int e = blockIdx.x * 8 + wave;
  if (e >= E) return;
  int s = src_idx[e];
  int d = dst_idx[e];
  float w = wts[e];
  const float4* sp = (const float4*)(nft + (size_t)s * 256) + lane * 2;
  float4 v0 = sp[0];
  float4 v1 = sp[1];
  float* dp = nacc + (size_t)d * 256 + lane * 8;
  atomicAdd(dp + 0, w * v0.x); atomicAdd(dp + 1, w * v0.y);
  atomicAdd(dp + 2, w * v0.z); atomicAdd(dp + 3, w * v0.w);
  atomicAdd(dp + 4, w * v1.x); atomicAdd(dp + 5, w * v1.y);
  atomicAdd(dp + 6, w * v1.z); atomicAdd(dp + 7, w * v1.w);
  if (lane == 0) atomicAdd(wsum + d, w);
}

// ---------- ws = clip(sum_w, 1, inf); store reciprocal ----------
__global__ void finalize_ws(const float* __restrict__ wsum,
                            float* __restrict__ invw, int n) {
  int i = blockIdx.x * blockDim.x + threadIdx.x;
  if (i < n) invw[i] = 1.f / fmaxf(wsum[i], 1.f);
}

// ---------- per-row L2 normalize (in place on d_out) ----------
__global__ __launch_bounds__(256)
void l2norm_rows(float* __restrict__ z) {
  __shared__ float red[256];
  int row = blockIdx.x;
  int t = threadIdx.x;
  float v = z[(size_t)row * 256 + t];
  red[t] = v * v;
  __syncthreads();
  for (int s = 128; s > 0; s >>= 1) {
    if (t < s) red[t] += red[t + s];
    __syncthreads();
  }
  float nrm = sqrtf(red[0]);
  float inv = (nrm == 0.f) ? 1.f : 1.f / nrm;
  z[(size_t)row * 256 + t] = v * inv;
}

extern "C" void kernel_launch(void* const* d_in, const int* in_sizes, int n_in,
                              void* d_out, int out_size, void* d_ws, size_t ws_size,
                              hipStream_t stream)
{
  const float* h_src   = (const float*)d_in[0];
  const float* h_dst   = (const float*)d_in[1];
  const float* wts     = (const float*)d_in[2];
  const float* Q_w     = (const float*)d_in[3];   // [256,256] row-major [in,out]
  const float* Q_b     = (const float*)d_in[4];
  const float* W_w     = (const float*)d_in[5];   // [512,256]
  const float* W_b     = (const float*)d_in[6];
  const int*   src_idx = (const int*)d_in[7];
  const int*   dst_idx = (const int*)d_in[8];
  float* out = (float*)d_out;

  // workspace layout (floats): nft | nacc | wsum | invw | Qt(bf16) | Wt(bf16)
  float* nft  = (float*)d_ws;
  float* nacc = nft  + (size_t)N_SRC * 256;
  float* wsum = nacc + (size_t)N_DST * 256;
  float* invw = wsum + N_DST;
  unsigned short* Qt = (unsigned short*)(invw + N_DST);   // 256*256 bf16
  unsigned short* Wt = Qt + 256 * 256;                    // 256*512 bf16

  // 0) pack weights to bf16 col-major (tiny, L2-resident)
  pack_b_bf16<<<dim3(1, 256), 256, 0, stream>>>(Q_w, Qt, 256);
  pack_b_bf16<<<dim3(2, 256), 256, 0, stream>>>(W_w, Wt, 512);

  // 1) zero scatter accumulators (must happen every call)
  {
    int n = N_DST * 256 + N_DST;             // nacc + wsum contiguous
    zero_f32<<<(n + 255) / 256, 256, 0, stream>>>(nacc, n);
  }
  // 2) nft = relu(h_src @ Q_w + Q_b)
  {
    dim3 grid((N_SRC + 63) / 64, 4);
    gemm_bf16_relu<256, 0><<<grid, 128, 0, stream>>>(h_src, nullptr, nullptr,
                                                     Qt, Q_b, nft, N_SRC);
  }
  // 3) weighted scatter over edges
  edge_scatter<<<(N_EDGE + 7) / 8, 256, 0, stream>>>(nft, wts, src_idx, dst_idx,
                                                     nacc, wsum, N_EDGE);
  // 4) invw = 1 / clip(wsum, 1, inf)
  finalize_ws<<<(N_DST + 255) / 256, 256, 0, stream>>>(wsum, invw, N_DST);
  // 5) out = relu(concat(nacc*invw, h_dst) @ W_w + W_b)
  {
    dim3 grid((N_DST + 63) / 64, 4);
    gemm_bf16_relu<512, 1><<<grid, 128, 0, stream>>>(nacc, h_dst, invw,
                                                     Wt, W_b, out, N_DST);
  }
  // 6) row-wise L2 normalization (in place)
  l2norm_rows<<<N_DST, 256, 0, stream>>>(out);
}
